// MultiHeadGraphConvLayer_90263032692982
// MI455X (gfx1250) — matrix operand order repaired and checked by reading, not compiled, not run on hardware
//
#include <hip/hip_runtime.h>
#include <hip/hip_bf16.h>

typedef __attribute__((ext_vector_type(16))) _Float16 v16h;
typedef __attribute__((ext_vector_type(8)))  _Float16 v8h;
typedef __attribute__((ext_vector_type(2)))  _Float16 h2;
typedef __attribute__((ext_vector_type(8)))  float    v8f;

// Problem constants
#define BB   32
#define NN   128
#define DD   128
#define BOND 16
#define HH   8
#define A2   128
#define OH   16
#define OO   128

// LDS layout for the fused kernel
#define LOGIT_STRIDE 129                       // floats, pad vs 64 banks
#define ATT_STRIDE   136                       // halves, 272B rows -> bank+4
#define LOGIT_BYTES  (16 * 8 * LOGIT_STRIDE * 4)   // 66048
#define ATT_BYTES    (8 * 16 * ATT_STRIDE * 2)     // 34816
#define SMEM_BYTES   (LOGIT_BYTES + ATT_BYTES)     // 100864 (~98.5 KB)

__device__ __forceinline__ v8f wmma_f16(v16h a, v16h b, v8f c) {
  return __builtin_amdgcn_wmma_f32_16x16x32_f16(false, a, false, b, (short)0, c,
                                                false, false);
}

__device__ __forceinline__ float leakyf(float z) {
  return fmaxf(z, 0.01f * z);
}

// A-fragment (16x32 f16, M x K): half-slot t (0..15) -> local k
// lanes 0-15: K 0..15 ; lanes 16-31: K 16..31
__device__ __forceinline__ int afrag_k(int t, int hi) {
  return hi * 16 + t;
}
// B-fragment (32x16 f16, K x N): lanes 0-15 hold K 0..15, lanes 16-31 K 16..31.

// packed-f16 leaky on 8 lanes: max(z, 0.01*z) == leaky_relu(z)
__device__ __forceinline__ v8h leaky8(v8h z) {
  return __builtin_elementwise_max(z, z * (_Float16)0.01f);
}

__device__ __forceinline__ v16h cat8(v8h a, v8h b) {
  return __builtin_shufflevector(a, b, 0, 1, 2, 3, 4, 5, 6, 7, 8, 9, 10, 11,
                                 12, 13, 14, 15);
}

// ---------------------------------------------------------------------------
// Kernel 1: rows_h = f16(x @ W_att[:D]); colsp_h = f16(x @ W_att[D:] + b_att)
// M=4096, N=128, K=128. grid 256 M-tiles, 8 waves each owning one a-tile.
// ---------------------------------------------------------------------------
__global__ void __launch_bounds__(256) precompute_rc(
    const float* __restrict__ x, const float* __restrict__ W_att,
    const float* __restrict__ b_att, _Float16* __restrict__ rows_h,
    _Float16* __restrict__ colsp_h) {
  int tile = blockIdx.x;
  int wave = threadIdx.x >> 5;
  int lane = threadIdx.x & 31;
  int mlane = lane & 15, hi = lane >> 4, n = lane & 15;
  int a0 = wave * 16;
  const float* xr = x + (tile * 16 + mlane) * DD;

  v8f accR = {}, accC = {};
#pragma unroll
  for (int ks = 0; ks < 4; ks++) {
    v16h af, br, bc;
#pragma unroll
    for (int t = 0; t < 16; t++) af[t] = (_Float16)xr[ks * 32 + afrag_k(t, hi)];
    int kb = ks * 32 + hi * 16;
#pragma unroll
    for (int t = 0; t < 16; t++) {
      br[t] = (_Float16)W_att[(kb + t) * A2 + a0 + n];        // W_att[:D]
      bc[t] = (_Float16)W_att[(DD + kb + t) * A2 + a0 + n];   // W_att[D:]
    }
    accR = wmma_f16(af, br, accR);
    accC = wmma_f16(af, bc, accC);
  }
  float ba = b_att[a0 + n];
#pragma unroll
  for (int v = 0; v < 8; v++) {
    int mm = tile * 16 + hi * 8 + v;
    rows_h[mm * A2 + a0 + n]  = (_Float16)accR[v];
    colsp_h[mm * A2 + a0 + n] = (_Float16)(accC[v] + ba);
  }
}

// ---------------------------------------------------------------------------
// Kernel 1b: xT[b][d][j] = f16(x[b][j][d])  (B-fragment friendly layout)
// ---------------------------------------------------------------------------
__global__ void __launch_bounds__(256) transpose_x(const float* __restrict__ x,
                                                   _Float16* __restrict__ xT) {
  int idx = blockIdx.x * 256 + threadIdx.x;
  if (idx >= BB * DD * NN) return;
  int j = idx & (NN - 1);
  int d = (idx >> 7) & (DD - 1);
  int b = idx >> 14;
  xT[(b * DD + d) * NN + j] = (_Float16)x[(b * NN + j) * DD + d];
}

// ---------------------------------------------------------------------------
// Fused kernel: logits (5 WMMAs/tile) -> softmax (att stays in LDS) ->
// agg = att @ x (WMMA) -> per-head conv (WMMA) + bias + residual + leaky.
// grid = B*(N/16) = 256 blocks, 256 threads (8 waves, ~6 waves/SIMD at
// 3 blocks/WGP). Phase 1: wave -> 2 i-rows; Phase 2: wave -> 1 head.
// ---------------------------------------------------------------------------
__global__ void __launch_bounds__(256) fused_attn_kernel(
    const _Float16* __restrict__ rows_h, const _Float16* __restrict__ colsp_h,
    const float* __restrict__ adj, const float* __restrict__ mask,
    const float* __restrict__ soft_mask, const float* __restrict__ W_fin,
    const float* __restrict__ b_fin, const _Float16* __restrict__ xT,
    const float* __restrict__ x, const float* __restrict__ W_conv,
    const float* __restrict__ b_conv, float* __restrict__ out) {
  __shared__ __align__(16) char smem[SMEM_BYTES];
  float* logits = (float*)smem;                         // [i][h][j] str 129
  _Float16* att_lds = (_Float16*)(smem + LOGIT_BYTES);  // [h][i][j] str 136
  float* aggbuf = (float*)smem;                         // aliases logits (ph.2)

  int b = blockIdx.x >> 3;
  int i0 = (blockIdx.x & 7) * 16;
  int wave = threadIdx.x >> 5, lane = threadIdx.x & 31;
  int mlane = lane & 15, hi = lane >> 4, n = lane & 15;

  // ---- Phase 1: logits --------------------------------------------------
  v16h wf[4], wadj;
#pragma unroll
  for (int ks = 0; ks < 4; ks++) {
    int kb = ks * 32 + hi * 16;
#pragma unroll
    for (int t = 0; t < 16; t++)
      wf[ks][t] = (n < HH) ? (_Float16)W_fin[(kb + t) * HH + n] : (_Float16)0.f;
  }
#pragma unroll
  for (int t = 0; t < 16; t++)
    wadj[t] = (hi == 0 && n < HH) ? (_Float16)W_fin[(A2 + t) * HH + n]
                                  : (_Float16)0.f;

  const _Float16* rowsb = rows_h + b * NN * A2;
  const _Float16* colsb = colsp_h + b * NN * A2;

  for (int ii = 0; ii < 2; ii++) {
    int il = wave * 2 + ii;                  // wave -> 2 i-rows
    int gi = i0 + il;
    const _Float16* crow = colsb + gi * A2;
#pragma unroll 1
    for (int jt = 0; jt < 8; jt++) {
      int j0 = jt * 16;
      int jrow = j0 + mlane;
      const _Float16* rrow = rowsb + jrow * A2;
      const float* arow = adj + (((b * NN + gi) * NN) + jrow) * BOND;
      if (jt < 7)  // prefetch next j-tile's adj stream (global_prefetch_b8)
        __builtin_prefetch(arow + 16 * BOND, 0, 0);

      v8f acc = {};
#pragma unroll
      for (int ks = 0; ks < 4; ks++) {
        int c1 = ks * 32 + hi * 16, c2 = c1 + 8;
        v8h z0 = *(const v8h*)(rrow + c1) + *(const v8h*)(crow + c1);
        v8h z1 = *(const v8h*)(rrow + c2) + *(const v8h*)(crow + c2);
        acc = wmma_f16(cat8(leaky8(z0), leaky8(z1)), wf[ks], acc);
      }
      {  // adj contribution: K=16 zero-padded to 32
        v16h aadj;
#pragma unroll
        for (int t = 0; t < 16; t++) {
          int k = afrag_k(t, hi);
          aadj[t] = (k < BOND) ? (_Float16)arow[k] : (_Float16)0.f;
        }
        acc = wmma_f16(aadj, wadj, acc);
      }
      if (n < HH) {
#pragma unroll
        for (int v = 0; v < 8; v++)
          logits[(il * 8 + n) * LOGIT_STRIDE + j0 + hi * 8 + v] = acc[v];
      }
    }
  }
  __syncthreads();

  // ---- Softmax over j per (i,h); att -> LDS f16 -------------------------
  if (threadIdx.x < 128) {
    int t = threadIdx.x;
    int il = t >> 3, h = t & 7;
    int gi = i0 + il;
    float* lrow = logits + (il * 8 + h) * LOGIT_STRIDE;
    const float* smrow = soft_mask + (b * NN + gi) * NN;
    const float* mrow = mask + (b * NN + gi) * NN;
    float bf = b_fin[h];
    float mx = -1e30f;
    for (int j = 0; j < NN; j++) {
      float v = lrow[j] + bf + smrow[j];
      lrow[j] = v;
      mx = fmaxf(mx, v);
    }
    float s = 0.f;
    for (int j = 0; j < NN; j++) {
      float e = __expf(lrow[j] - mx);
      lrow[j] = e;
      s += e;
    }
    float inv = 1.f / s;
    _Float16* aout = att_lds + (h * 16 + il) * ATT_STRIDE;
    for (int j = 0; j < NN; j += 2) {
      h2 p;
      p[0] = (_Float16)(lrow[j] * inv * mrow[j]);
      p[1] = (_Float16)(lrow[j + 1] * inv * mrow[j + 1]);
      *(h2*)(aout + j) = p;
    }
  }
  __syncthreads();   // logits region now dead; aggbuf may alias it

  // ---- Phase 2: agg = att @ x, then conv + bias + residual + leaky ------
  // wave -> head; 8 waves x 8KB agg = 64KB, fits inside dead logits region
  {
    int h = wave;
    float* agg = aggbuf + wave * (16 * DD);

#pragma unroll 1
    for (int dt = 0; dt < 8; dt++) {
      int d0 = dt * 16;
      v8f acc = {};
#pragma unroll
      for (int ks = 0; ks < 4; ks++) {
        const _Float16* arow =
            att_lds + (h * 16 + mlane) * ATT_STRIDE + ks * 32 + hi * 16;
        v16h af = cat8(*(const v8h*)(arow), *(const v8h*)(arow + 8));
        v16h bf =
            *(const v16h*)(xT + (b * DD + d0 + n) * NN + ks * 32 + hi * 16);
        acc = wmma_f16(af, bf, acc);
      }
#pragma unroll
      for (int v = 0; v < 8; v++) agg[(hi * 8 + v) * DD + d0 + n] = acc[v];
    }

    // conv: out16x16 = agg(16x128) @ W_conv[h](128x16)
    v8f acc2 = {};
#pragma unroll
    for (int ks = 0; ks < 4; ks++) {
      v16h af, bf;
      const float* grow = agg + mlane * DD + ks * 32;
#pragma unroll
      for (int t = 0; t < 16; t++) af[t] = (_Float16)grow[afrag_k(t, hi)];
      const float* wrow = W_conv + (h * DD + ks * 32 + hi * 16) * OH + n;
#pragma unroll
      for (int t = 0; t < 16; t++) bf[t] = (_Float16)wrow[t * OH];
      acc2 = wmma_f16(af, bf, acc2);
    }
    float bc = b_conv[h * OH + n];
#pragma unroll
    for (int v = 0; v < 8; v++) {
      int gi = i0 + hi * 8 + v;
      int c = h * OH + n;
      float z = acc2[v] + bc + x[(b * NN + gi) * DD + c];
      out[(b * NN + gi) * OO + c] = leakyf(z);
    }
  }
}

// ---------------------------------------------------------------------------
extern "C" void kernel_launch(void* const* d_in, const int* in_sizes, int n_in,
                              void* d_out, int out_size, void* d_ws,
                              size_t ws_size, hipStream_t stream) {
  const float* x         = (const float*)d_in[0];
  const float* adj       = (const float*)d_in[1];
  const float* mask      = (const float*)d_in[2];
  const float* soft_mask = (const float*)d_in[3];
  const float* W_att     = (const float*)d_in[4];
  const float* b_att     = (const float*)d_in[5];
  const float* W_fin     = (const float*)d_in[6];
  const float* b_fin     = (const float*)d_in[7];
  const float* W_conv    = (const float*)d_in[8];
  const float* b_conv    = (const float*)d_in[9];
  float* out = (float*)d_out;

  char* ws = (char*)d_ws;
  _Float16* rows_h  = (_Float16*)ws;                       // 1 MB
  _Float16* colsp_h = (_Float16*)(ws + (1u << 20));        // 1 MB
  _Float16* xT      = (_Float16*)(ws + (2u << 20));        // 1 MB

  precompute_rc<<<256, 256, 0, stream>>>(x, W_att, b_att, rows_h, colsp_h);
  transpose_x<<<(BB * DD * NN + 255) / 256, 256, 0, stream>>>(x, xT);
  fused_attn_kernel<<<BB * (NN / 16), 256, 0, stream>>>(
      rows_h, colsp_h, adj, mask, soft_mask, W_fin, b_fin, xT, x, W_conv,
      b_conv, out);
}
